// DeepSetElectionModel_6571299962914
// MI455X (gfx1250) — compile-verified, matrix operand order checked by reading, not run on hardware
//
#include <hip/hip_runtime.h>
#include <hip/hip_bf16.h>
#include <math.h>

typedef __attribute__((ext_vector_type(16))) __bf16 v16bf;
typedef __attribute__((ext_vector_type(8)))  float  v8f;

#define NV      1048576
#define NC      32
#define EMBD    128
#define NSEG    4096
#define NTILES  (NV / 16)
#define WPB     8          // waves per block (256 threads, wave32)
#define BLOCKS1 1024

// LDS layout for kernel 1:
//  - wfrag: 72 WMMA B-fragments (bf16, hardware register layout), 1KB each
//      frag 0..7   : layer1 (colTile ct, kt=0)        B = lW1[32x128]
//      frag 8..39  : layer2 (ct*4+kt)                 B = lW2[128x128]
//      frag 40..71 : layer3 (ct*4+kt)                 B = lW3[128x128]
//  - bounce: per-wave 16x136 bf16 tile (pitch padded for banks; 272B rows keep
//      all A-fragment ds_load_b128 16B-aligned)
//  - bias: lb1 | lb2 | lb3
struct SmemK1 {
  uint4 wfrag[72 * 64];
  uint4 bounce[WPB * 272];
  float bias[3 * EMBD];
};

__device__ __forceinline__ v16bf load_wfrag(const uint4* wf, int f, int lane) {
  union { uint4 u[2]; v16bf v; } c;
  const uint4* p = wf + f * 64 + lane * 2;   // 32B per lane, contiguous
  c.u[0] = p[0];
  c.u[1] = p[1];
  return c.v;
}

// One K=128 layer: out[ct] = A(16x128) x B(128x16ct), B fragments streamed
// from LDS with one-col-tile-ahead preload so ds waits stagger instead of
// serializing every WMMA.
__device__ __forceinline__ void layer_k128(const uint4* wf, int fbase, int lane,
                                           const v16bf a[4], v8f acc[8]) {
  v16bf bb[2][4];
#pragma unroll
  for (int kt = 0; kt < 4; ++kt) bb[0][kt] = load_wfrag(wf, fbase + kt, lane);
#pragma unroll
  for (int ct = 0; ct < 8; ++ct) {
    const int cur = ct & 1, nxt = cur ^ 1;
    if (ct < 7) {
#pragma unroll
      for (int kt = 0; kt < 4; ++kt)
        bb[nxt][kt] = load_wfrag(wf, fbase + (ct + 1) * 4 + kt, lane);
    }
    v8f c = {0.f, 0.f, 0.f, 0.f, 0.f, 0.f, 0.f, 0.f};
#pragma unroll
    for (int kt = 0; kt < 4; ++kt)
      c = __builtin_amdgcn_wmma_f32_16x16x32_bf16(false, a[kt], false, bb[cur][kt],
                                                  (short)0, c, false, false);
    acc[ct] = c;
  }
}

__global__ __launch_bounds__(256) void local_mlp_scatter(
    const float* __restrict__ x, const int* __restrict__ seg,
    const float* __restrict__ lW1, const float* __restrict__ lb1,
    const float* __restrict__ lW2, const float* __restrict__ lb2,
    const float* __restrict__ lW3, const float* __restrict__ lb3,
    float* __restrict__ agg)
{
  __shared__ SmemK1 sm;
  const int tid = threadIdx.x;

  // ---- one-time per-block: stage weights as bf16 WMMA-B fragments ----
  for (int idx = tid; idx < 72 * 512; idx += 256) {
    const int f   = idx >> 9;
    const int rem = idx & 511;
    const int l   = rem >> 4;        // lane 0..31
    const int i   = rem & 15;        // half index within lane
    const int h2  = l >> 4;
    int ct, kt;
    const float* w;
    if (f < 8)       { ct = f;        kt = 0;              w = lW1; }
    else if (f < 40) { int g = f - 8;  ct = g >> 2; kt = g & 3; w = lW2; }
    else             { int g = f - 40; ct = g >> 2; kt = g & 3; w = lW3; }
    const int kk = kt * 32 + h2 * 8 + (i < 8 ? i : i + 8);  // ISA 16-bit B layout
    const int nn = ct * 16 + (l & 15);
    ((__bf16*)sm.wfrag)[idx] = (__bf16)w[kk * EMBD + nn];
  }
  for (int j = tid; j < 3 * EMBD; j += 256)
    sm.bias[j] = (j < EMBD) ? lb1[j] : (j < 2 * EMBD) ? lb2[j - EMBD] : lb3[j - 2 * EMBD];
  __syncthreads();

  const int wave = tid >> 5, lane = tid & 31, ln = lane & 15, hi = lane >> 4;
  __bf16* bounce = (__bf16*)(sm.bounce + wave * 272);   // 16 rows x 136 halfs
  const float* biasL1 = sm.bias;
  const float* biasL2 = sm.bias + EMBD;
  const float* biasL3 = sm.bias + 2 * EMBD;

  const int gw = blockIdx.x * WPB + wave;
  for (int tile = gw; tile < NTILES; tile += BLOCKS1 * WPB) {
    const int r0 = tile * 16;

    // ---- segment ids for this lane's 8 C-layout rows (hoisted: latency
    //      hides behind the 72-WMMA body) ----
    const int4 sv0 = *(const int4*)(seg + r0 + hi * 8);
    const int4 sv1 = *(const int4*)(seg + r0 + hi * 8 + 4);
    int s[8];
    s[0] = sv0.x; s[1] = sv0.y; s[2] = sv0.z; s[3] = sv0.w;
    s[4] = sv1.x; s[5] = sv1.y; s[6] = sv1.z; s[7] = sv1.w;

    // ---- load x tile (16x32) as WMMA A fragment, f32 -> bf16 ----
    const float* xr = x + (size_t)(r0 + ln) * NC + hi * 8;
    float4 q0 = ((const float4*)xr)[0];
    float4 q1 = ((const float4*)xr)[1];
    float4 q2 = ((const float4*)(xr + 16))[0];
    float4 q3 = ((const float4*)(xr + 16))[1];
    v16bf a1;
    a1[0]  = (__bf16)q0.x; a1[1]  = (__bf16)q0.y; a1[2]  = (__bf16)q0.z; a1[3]  = (__bf16)q0.w;
    a1[4]  = (__bf16)q1.x; a1[5]  = (__bf16)q1.y; a1[6]  = (__bf16)q1.z; a1[7]  = (__bf16)q1.w;
    a1[8]  = (__bf16)q2.x; a1[9]  = (__bf16)q2.y; a1[10] = (__bf16)q2.z; a1[11] = (__bf16)q2.w;
    a1[12] = (__bf16)q3.x; a1[13] = (__bf16)q3.y; a1[14] = (__bf16)q3.z; a1[15] = (__bf16)q3.w;

    v8f acc[8];

    // ---- layer 1: h1 = relu(x @ lW1 + lb1), K = 32 ----
    // preload all 8 B fragments into distinct regs so ds waits stagger
    {
      v16bf b1r[8];
#pragma unroll
      for (int ct = 0; ct < 8; ++ct) b1r[ct] = load_wfrag(sm.wfrag, ct, lane);
#pragma unroll
      for (int ct = 0; ct < 8; ++ct) {
        v8f z = {0.f, 0.f, 0.f, 0.f, 0.f, 0.f, 0.f, 0.f};
        acc[ct] = __builtin_amdgcn_wmma_f32_16x16x32_bf16(false, a1, false, b1r[ct],
                                                          (short)0, z, false, false);
      }
    }
#pragma unroll
    for (int ct = 0; ct < 8; ++ct) {
      const float bn = biasL1[ct * 16 + ln];
#pragma unroll
      for (int v = 0; v < 8; ++v) {
        float t = acc[ct][v] + bn;
        t = t > 0.f ? t : 0.f;
        bounce[(hi * 8 + v) * 136 + ct * 16 + ln] = (__bf16)t;
      }
    }

    // ---- layer 2: K = 128 ----
    v16bf a2[4];
#pragma unroll
    for (int kt = 0; kt < 4; ++kt) {
      union { uint4 u[2]; v16bf v; } c;
      const char* p = (const char*)bounce + ln * 272 + kt * 64 + hi * 16;
      c.u[0] = *(const uint4*)p;
      c.u[1] = *(const uint4*)(p + 32);
      a2[kt] = c.v;
    }
    layer_k128(sm.wfrag, 8, lane, a2, acc);
#pragma unroll
    for (int ct = 0; ct < 8; ++ct) {
      const float bn = biasL2[ct * 16 + ln];
#pragma unroll
      for (int v = 0; v < 8; ++v) {
        float t = acc[ct][v] + bn;
        t = t > 0.f ? t : 0.f;
        bounce[(hi * 8 + v) * 136 + ct * 16 + ln] = (__bf16)t;
      }
    }

    // ---- layer 3: K = 128, no relu ----
    v16bf a3[4];
#pragma unroll
    for (int kt = 0; kt < 4; ++kt) {
      union { uint4 u[2]; v16bf v; } c;
      const char* p = (const char*)bounce + ln * 272 + kt * 64 + hi * 16;
      c.u[0] = *(const uint4*)p;
      c.u[1] = *(const uint4*)(p + 32);
      a3[kt] = c.v;
    }
    layer_k128(sm.wfrag, 40, lane, a3, acc);

    // ---- sorted-segment scatter-sum: run-length accumulate then atomic ----
#pragma unroll
    for (int ct = 0; ct < 8; ++ct) {
      const int   n  = ct * 16 + ln;
      const float bn = biasL3[n];
      int   cur = s[0];
      float run = acc[ct][0] + bn;
#pragma unroll
      for (int v = 1; v < 8; ++v) {
        const float t = acc[ct][v] + bn;
        if (s[v] == cur) {
          run += t;
        } else {
          unsafeAtomicAdd(&agg[(size_t)cur * EMBD + n], run);
          cur = s[v];
          run = t;
        }
      }
      unsafeAtomicAdd(&agg[(size_t)cur * EMBD + n], run);
    }
  }
}

// ---- kernel 2: global MLP + log_softmax, exact fp32, 16 segments/block ----
__global__ __launch_bounds__(128) void global_mlp_softmax(
    const float* __restrict__ agg,
    const float* __restrict__ gW1, const float* __restrict__ gb1,
    const float* __restrict__ gW2, const float* __restrict__ gb2,
    const float* __restrict__ gW3, const float* __restrict__ gb3,
    float* __restrict__ out)
{
  __shared__ float sA[16][EMBD];
  __shared__ float sB[16][EMBD];
  const int t  = threadIdx.x;
  const int s0 = blockIdx.x * 16;

  for (int i = t; i < 16 * EMBD; i += 128)
    sA[i >> 7][i & 127] = agg[(size_t)(s0 + (i >> 7)) * EMBD + (i & 127)];
  __syncthreads();

  float acc[16];
#pragma unroll
  for (int r = 0; r < 16; ++r) acc[r] = 0.f;
  for (int k = 0; k < EMBD; ++k) {
    const float w = gW1[k * EMBD + t];          // coalesced; sA[r][k] broadcast
#pragma unroll
    for (int r = 0; r < 16; ++r) acc[r] = fmaf(sA[r][k], w, acc[r]);
  }
  const float b1 = gb1[t];
#pragma unroll
  for (int r = 0; r < 16; ++r) { float v = acc[r] + b1; sB[r][t] = v > 0.f ? v : 0.f; }
  __syncthreads();

#pragma unroll
  for (int r = 0; r < 16; ++r) acc[r] = 0.f;
  for (int k = 0; k < EMBD; ++k) {
    const float w = gW2[k * EMBD + t];
#pragma unroll
    for (int r = 0; r < 16; ++r) acc[r] = fmaf(sB[r][k], w, acc[r]);
  }
  const float b2 = gb2[t];
#pragma unroll
  for (int r = 0; r < 16; ++r) { float v = acc[r] + b2; sA[r][t] = v > 0.f ? v : 0.f; }
  __syncthreads();

  // scores + log_softmax: one wave32 per row (c = lane), 4 passes of 4 rows
  const int r4 = t >> 5, c = t & 31;
  for (int p = 0; p < 4; ++p) {
    const int r = p * 4 + r4;
    float sc = gb3[c];
    for (int k = 0; k < EMBD; ++k) sc = fmaf(sA[r][k], gW3[k * NC + c], sc);
    float mx = sc;
#pragma unroll
    for (int off = 16; off > 0; off >>= 1) mx = fmaxf(mx, __shfl_xor(mx, off, 32));
    const float e = expf(sc - mx);
    float sum = e;
#pragma unroll
    for (int off = 16; off > 0; off >>= 1) sum += __shfl_xor(sum, off, 32);
    out[(size_t)(s0 + r) * NC + c] = sc - mx - logf(sum);
  }
}

extern "C" void kernel_launch(void* const* d_in, const int* in_sizes, int n_in,
                              void* d_out, int out_size, void* d_ws, size_t ws_size,
                              hipStream_t stream) {
  const float* x   = (const float*)d_in[0];
  const int*   idx = (const int*)  d_in[1];
  const float* lW1 = (const float*)d_in[2];  const float* lb1 = (const float*)d_in[3];
  const float* lW2 = (const float*)d_in[4];  const float* lb2 = (const float*)d_in[5];
  const float* lW3 = (const float*)d_in[6];  const float* lb3 = (const float*)d_in[7];
  const float* gW1 = (const float*)d_in[8];  const float* gb1 = (const float*)d_in[9];
  const float* gW2 = (const float*)d_in[10]; const float* gb2 = (const float*)d_in[11];
  const float* gW3 = (const float*)d_in[12]; const float* gb3 = (const float*)d_in[13];

  float* agg = (float*)d_ws;  // [NSEG][EMBD] accumulator
  hipMemsetAsync(agg, 0, (size_t)NSEG * EMBD * sizeof(float), stream);

  local_mlp_scatter<<<BLOCKS1, 256, 0, stream>>>(x, idx, lW1, lb1, lW2, lb2, lW3, lb3, agg);
  global_mlp_softmax<<<NSEG / 16, 128, 0, stream>>>(agg, gW1, gb1, gW2, gb2, gW3, gb3,
                                                    (float*)d_out);
}
